// EGCN_80401787781187
// MI455X (gfx1250) — matrix operand order repaired and checked by reading, not compiled, not run on hardware
//
#include <hip/hip_runtime.h>
#include <hip/hip_bf16.h>

// ---------------------------------------------------------------------------
// EGCN forward for MI455X (gfx1250, wave32).
//   - GEMMs: v_wmma_f32_16x16x32_bf16, f32 accumulate. Weights pre-packed to
//     a contiguous-k B-fragment lane layout (k = 16*(L/16) + i, same as A) and held
//     REGISTER-RESIDENT (2 n-tiles per wave, persistent grid-stride over row
//     strips) so the inner loop is pure global_load->cvt->wmma, no LDS stalls.
//   - Aggregation: CSR built once per launch (int-atomic degree count, tiled
//     exclusive scan, cursor placement), then a wave-per-node register gather
//     fused with self-loop + bias + SELU. Zero f32 atomics in the hot path;
//     activations (51 MB) stay L2-resident (192 MB L2).
// ---------------------------------------------------------------------------

#define NN 100000L   // nodes
#define NE 1600000L  // edges
#define NG 2048L     // graphs
#define DH 128       // hidden dim (= D_IN = D_EMB)
#define DM 64        // mol feats
#define DC 192       // concat dim
#define SCAN_TILE 1024

typedef __attribute__((ext_vector_type(16))) __bf16 v16bf;
typedef __attribute__((ext_vector_type(8)))  float  v8f;

__device__ __forceinline__ float selu_f(float x) {
  const float alpha = 1.6732632423543772f, scale = 1.0507009873554805f;
  return scale * (x > 0.0f ? x : alpha * (expf(x) - 1.0f));
}

// ---------------------------------------------------------------- utilities
__global__ __launch_bounds__(256) void fill_f(float* p, float v, long n) {
  long i = (long)blockIdx.x * 256 + threadIdx.x;
  if (i < n) p[i] = v;
}

__global__ __launch_bounds__(256) void fill_i(int* p, int v, long n) {
  long i = (long)blockIdx.x * 256 + threadIdx.x;
  if (i < n) p[i] = v;
}

// ------------------------------------------------------------- degree / CSR
__global__ __launch_bounds__(256) void edge_count(const int* __restrict__ dst,
                                                  int* __restrict__ cnt, int E) {
  int e = blockIdx.x * 256 + threadIdx.x;
  if (e < E) atomicAdd(&cnt[dst[e]], 1);
}

// dinv[i] = rsqrt(cnt[i] + 1)   (self loop)
__global__ __launch_bounds__(256) void dinv_from_cnt(const int* __restrict__ cnt,
                                                     float* __restrict__ dinv, long n) {
  long i = (long)blockIdx.x * 256 + threadIdx.x;
  if (i < n) dinv[i] = rsqrtf((float)cnt[i] + 1.0f);
}

// tile-level exclusive scan: 256 threads x 4 items = 1024/tile
__global__ __launch_bounds__(256) void scan_tile_k(const int* __restrict__ cnt,
                                                   int* __restrict__ rowptr,
                                                   int* __restrict__ tileSums, int n) {
  __shared__ int sh[256];
  const int t = threadIdx.x;
  const int base = blockIdx.x * SCAN_TILE + t * 4;
  int v0 = 0, v1 = 0, v2 = 0, v3 = 0;
  if (base + 0 < n) v0 = cnt[base + 0];
  if (base + 1 < n) v1 = cnt[base + 1];
  if (base + 2 < n) v2 = cnt[base + 2];
  if (base + 3 < n) v3 = cnt[base + 3];
  const int s = v0 + v1 + v2 + v3;
  sh[t] = s;
  __syncthreads();
  for (int off = 1; off < 256; off <<= 1) {   // Hillis-Steele inclusive
    int x = (t >= off) ? sh[t - off] : 0;
    __syncthreads();
    sh[t] += x;
    __syncthreads();
  }
  const int excl = sh[t] - s;
  if (base + 0 < n) rowptr[base + 0] = excl;
  if (base + 1 < n) rowptr[base + 1] = excl + v0;
  if (base + 2 < n) rowptr[base + 2] = excl + v0 + v1;
  if (base + 3 < n) rowptr[base + 3] = excl + v0 + v1 + v2;
  if (t == 255) tileSums[blockIdx.x] = sh[255];
}

__global__ void scan_aux_k(int* tileSums, int nt) {
  if (threadIdx.x == 0 && blockIdx.x == 0) {
    int run = 0;
    for (int i = 0; i < nt; ++i) { int v = tileSums[i]; tileSums[i] = run; run += v; }
  }
}

// add tile offsets, init cursor, terminate rowptr
__global__ __launch_bounds__(256) void scan_add_k(int* __restrict__ rowptr,
                                                  const int* __restrict__ tileSums,
                                                  int* __restrict__ cursor,
                                                  int n, int totalE) {
  long i = (long)blockIdx.x * 256 + threadIdx.x;
  if (i >= n) return;
  int r = rowptr[i] + tileSums[i >> 10];
  rowptr[i] = r;
  cursor[i] = r;
  if (i == 0) rowptr[n] = totalE;
}

// place edges into CSR, precompute normalization coefficient
__global__ __launch_bounds__(256) void edge_place(const int* __restrict__ srcI,
                                                  const int* __restrict__ dstI,
                                                  const float* __restrict__ dinv,
                                                  int* __restrict__ cursor,
                                                  int* __restrict__ csr_src,
                                                  float* __restrict__ csr_w, int E) {
  int e = blockIdx.x * 256 + threadIdx.x;
  if (e >= E) return;
  int s = srcI[e], d = dstI[e];
  int pos = atomicAdd(&cursor[d], 1);
  csr_src[pos] = s;
  csr_w[pos] = dinv[s] * dinv[d];
}

// ----------------------------------------------------- weight pre-packing
// Pack fp32 row-major W[K,128] into bf16 WMMA B-fragments:
// lane L holds col N = L%16; frag elem i holds K = (L/16)*16 + i in 32-K tile.
__global__ __launch_bounds__(256) void pack_w(const float* __restrict__ W,
                                              __bf16* __restrict__ out, int K) {
  int idx = blockIdx.x * 256 + threadIdx.x;
  if (idx >= K * 128) return;
  int i    = idx & 15;
  int lane = (idx >> 4) & 31;
  int t    = idx >> 9;       // kt*8 + nt
  int nt   = t & 7;
  int kt   = t >> 3;
  int k = kt * 32 + (lane >> 4) * 16 + i;
  int n = nt * 16 + (lane & 15);
  out[idx] = (__bf16)W[k * 128 + n];
}

// ----------------------------------------------------------- WMMA GEMM
// C[M,128] = A[M,K] @ W (pre-packed bf16). 8 waves/block; wave owns 2 n-tiles
// with ALL its B fragments in registers, grid-strides over 16-row strips.
// A frag (16x32 bf16): lane L row M=L%16; frag[0..15]=K((L/16)*16+0..15).
// D f32 16x16: frag[j] row (L/16)*8+j, col L%16.
template <int K>
__global__ __launch_bounds__(256) void gemm_wmma(const float* __restrict__ A,
                                                 const __bf16* __restrict__ Bp,
                                                 float* __restrict__ C, long M) {
  constexpr int KT = K / 32;
  const int tid  = threadIdx.x;
  const int wave = tid >> 5;
  const int lane = tid & 31;
  const int half = lane >> 4;
  const int l16  = lane & 15;
  const int np   = wave & 3;    // n-tile pair: tiles 2*np, 2*np+1
  const int wrow = wave >> 2;   // row-strip slot within block (0/1)

  // register-resident B fragments (one 32B load each, stays live)
  v16bf bf[KT][2];
#pragma unroll
  for (int kt = 0; kt < KT; ++kt) {
#pragma unroll
    for (int j = 0; j < 2; ++j)
      bf[kt][j] = *(const v16bf*)&Bp[((kt * 8 + (np * 2 + j)) * 32 + lane) * 16];
  }

  const long strips = (M + 15) >> 4;
  const long step   = (long)gridDim.x * 2;
  for (long s = (long)blockIdx.x * 2 + wrow; s < strips; s += step) {
    const long row0 = s * 16;
    const float* arow = A + (row0 + l16) * K;
    long sn = s + step;
    if (sn < strips)
      __builtin_prefetch(A + (sn * 16 + l16) * K, 0, 1);  // global_prefetch_b8

    v8f acc0 = {}, acc1 = {};
#pragma unroll
    for (int kt = 0; kt < KT; ++kt) {
      v16bf af;
      const float* p0 = arow + kt * 32 + half * 16;
      const float* p1 = p0 + 8;
#pragma unroll
      for (int i = 0; i < 8; ++i) af[i] = (__bf16)p0[i];
#pragma unroll
      for (int i = 0; i < 8; ++i) af[8 + i] = (__bf16)p1[i];
      acc0 = __builtin_amdgcn_wmma_f32_16x16x32_bf16(false, af, false, bf[kt][0],
                                                     (short)0, acc0, false, false);
      acc1 = __builtin_amdgcn_wmma_f32_16x16x32_bf16(false, af, false, bf[kt][1],
                                                     (short)0, acc1, false, false);
    }
    const int n0 = np * 32 + l16;
#pragma unroll
    for (int j = 0; j < 8; ++j) {
      long m = row0 + half * 8 + j;
      C[m * DH + n0]      = acc0[j];
      C[m * DH + n0 + 16] = acc1[j];
    }
  }
}

// -------------------------------------------- fused CSR gather + bias + SELU
// one wave per node: h[i] = selu( xw[i]*dinv[i]^2 + sum_e xw[src_e]*w_e + b )
__global__ __launch_bounds__(256) void aggregate_fused(const float* __restrict__ xw,
                                                       const int* __restrict__ rowptr,
                                                       const int* __restrict__ csr_src,
                                                       const float* __restrict__ csr_w,
                                                       const float* __restrict__ dinv,
                                                       const float* __restrict__ bias,
                                                       float* __restrict__ hout, long n) {
  long t = (long)blockIdx.x * 256 + threadIdx.x;
  long i = t >> 5;
  if (i >= n) return;
  const int lane = t & 31;

  const float di = dinv[i];
  const float self = di * di;
  float4 v = ((const float4*)(xw + i * DH))[lane];
  float ax = v.x * self, ay = v.y * self, az = v.z * self, aw = v.w * self;

  const int e1 = rowptr[i + 1];
  for (int e = rowptr[i]; e < e1; ++e) {
    int s = csr_src[e];
    float w = csr_w[e];
    float4 u = ((const float4*)(xw + (long)s * DH))[lane];
    ax += u.x * w; ay += u.y * w; az += u.z * w; aw += u.w * w;
  }
  const float4 b4 = ((const float4*)bias)[lane];
  float4 r;
  r.x = selu_f(ax + b4.x);
  r.y = selu_f(ay + b4.y);
  r.z = selu_f(az + b4.z);
  r.w = selu_f(aw + b4.w);
  ((float4*)(hout + i * DH))[lane] = r;
}

// ----------------------------------------------------------- BatchNorm
__global__ __launch_bounds__(256) void bn_stats(const float* __restrict__ h,
                                                float* __restrict__ stats, long n) {
  __shared__ float ssum[256], ssq[256];
  int c = blockIdx.x;
  float s = 0.0f, q = 0.0f;
  for (long r = threadIdx.x; r < n; r += 256) {
    float v = h[r * DH + c];
    s += v; q += v * v;
  }
  ssum[threadIdx.x] = s; ssq[threadIdx.x] = q;
  __syncthreads();
  for (int off = 128; off > 0; off >>= 1) {
    if ((int)threadIdx.x < off) {
      ssum[threadIdx.x] += ssum[threadIdx.x + off];
      ssq[threadIdx.x]  += ssq[threadIdx.x + off];
    }
    __syncthreads();
  }
  if (threadIdx.x == 0) {
    float mean = ssum[0] / (float)n;
    stats[c]      = mean;
    stats[DH + c] = ssq[0] / (float)n - mean * mean;
  }
}

__global__ __launch_bounds__(256) void bn_apply(float* __restrict__ h,
                                                const float* __restrict__ stats,
                                                const float* __restrict__ g,
                                                const float* __restrict__ be,
                                                long total) {
  long idx = (long)blockIdx.x * 256 + threadIdx.x;
  if (idx >= total) return;
  int c = idx & (DH - 1);
  float inv = rsqrtf(stats[DH + c] + 1e-5f);
  h[idx] = (h[idx] - stats[c]) * inv * g[c] + be[c];
}

// bias + SELU for the head embedding
__global__ __launch_bounds__(256) void bias_selu(float* __restrict__ h,
                                                 const float* __restrict__ b,
                                                 long total) {
  long idx = (long)blockIdx.x * 256 + threadIdx.x;
  if (idx >= total) return;
  h[idx] = selu_f(h[idx] + b[idx & (DH - 1)]);
}

// --------------------------------------------------------------- pool / head
__global__ __launch_bounds__(256) void pool_add(const float* __restrict__ h3,
                                                const int* __restrict__ batch,
                                                float* __restrict__ hg, long n) {
  long t = (long)blockIdx.x * 256 + threadIdx.x;
  long i = t >> 5;
  if (i >= n) return;
  int lane = t & 31;
  int gidx = batch[i];
  const float4 v = ((const float4*)(h3 + i * DH))[lane];
  float* p = hg + (long)gidx * DH + lane * 4;
  atomicAdd(p + 0, v.x);
  atomicAdd(p + 1, v.y);
  atomicAdd(p + 2, v.z);
  atomicAdd(p + 3, v.w);
}

__global__ __launch_bounds__(256) void concat_k(const float* __restrict__ hg,
                                                const float* __restrict__ mol,
                                                float* __restrict__ hcat) {
  int idx = blockIdx.x * 256 + threadIdx.x;
  if (idx >= (int)(NG * DC)) return;
  int gidx = idx / DC, c = idx % DC;
  hcat[idx] = (c < DH) ? hg[gidx * DH + c] : mol[gidx * DM + (c - DH)];
}

__global__ __launch_bounds__(64) void out_dot(const float* __restrict__ emb,
                                              const float* __restrict__ Wf2,
                                              const float* __restrict__ bf2,
                                              float* __restrict__ out) {
  int row = blockIdx.x * 2 + (threadIdx.x >> 5);
  int lane = threadIdx.x & 31;
  const float4 v = ((const float4*)(emb + (long)row * DH))[lane];
  const float4 w = ((const float4*)Wf2)[lane];
  float s = v.x * w.x + v.y * w.y + v.z * w.z + v.w * w.w;
  for (int off = 16; off > 0; off >>= 1) s += __shfl_down(s, off, 32);
  if (lane == 0) out[row] = s + bf2[0];
}

// ---------------------------------------------------------------------------
extern "C" void kernel_launch(void* const* d_in, const int* in_sizes, int n_in,
                              void* d_out, int out_size, void* d_ws, size_t ws_size,
                              hipStream_t stream) {
  (void)in_sizes; (void)n_in; (void)out_size; (void)ws_size;
  const float* x    = (const float*)d_in[0];
  const int*   ei   = (const int*)d_in[1];   // [2, NE]
  const int*   batch= (const int*)d_in[2];
  const float* mol  = (const float*)d_in[3];
  const float* W1 = (const float*)d_in[4];  const float* b1 = (const float*)d_in[5];
  const float* g1 = (const float*)d_in[6];  const float* be1= (const float*)d_in[7];
  const float* W2 = (const float*)d_in[8];  const float* b2 = (const float*)d_in[9];
  const float* g2 = (const float*)d_in[10]; const float* be2= (const float*)d_in[11];
  const float* W3 = (const float*)d_in[12]; const float* b3 = (const float*)d_in[13];
  const float* Wf1= (const float*)d_in[14]; const float* bf1= (const float*)d_in[15];
  const float* Wf2= (const float*)d_in[16]; const float* bf2= (const float*)d_in[17];
  float* out = (float*)d_out;

  const int* srcI = ei;
  const int* dstI = ei + NE;

  // workspace carving (256B aligned)
  char* ws = (char*)d_ws;
  size_t off = 0;
  auto carve = [&](size_t bytes) -> size_t {
    size_t p = off; off = p + ((bytes + 255) & ~(size_t)255); return p;
  };
  float*  dinv    = (float*)(ws + carve(NN * 4));
  int*    cnt     = (int*)  (ws + carve(NN * 4));
  int*    rowptr  = (int*)  (ws + carve((NN + 1) * 4));
  int*    cursor  = (int*)  (ws + carve(NN * 4));
  int*    tileSum = (int*)  (ws + carve(256 * 4));
  int*    csr_src = (int*)  (ws + carve(NE * 4));
  float*  csr_w   = (float*)(ws + carve(NE * 4));
  float*  bufA    = (float*)(ws + carve(NN * DH * 4));
  float*  bufB    = (float*)(ws + carve(NN * DH * 4));
  float*  bufC    = (float*)(ws + carve(NN * DH * 4));
  __bf16* Wp1     = (__bf16*)(ws + carve(DH * DH * 2));
  __bf16* Wp2     = (__bf16*)(ws + carve(DH * DH * 2));
  __bf16* Wp3     = (__bf16*)(ws + carve(DH * DH * 2));
  __bf16* Wpf1    = (__bf16*)(ws + carve(DC * DH * 2));
  float*  stats   = (float*)(ws + carve(2 * DH * 4));
  float*  hg      = (float*)(ws + carve(NG * DH * 4));
  float*  hcat    = (float*)(ws + carve(NG * DC * 4));
  float*  emb     = (float*)(ws + carve(NG * DH * 4));

  const long totN  = NN * DH;
  const int gN     = (int)((totN + 255) / 256);
  const int gNode  = (int)((NN + 255) / 256);
  const int gEdge  = (int)((NE + 255) / 256);
  const int gAgg   = (int)((NN * 32 + 255) / 256);
  const int nTiles = (int)((NN + SCAN_TILE - 1) / SCAN_TILE);

  // ---- degree / dinv / CSR (graph fixed across all 3 layers)
  fill_i<<<gNode, 256, 0, stream>>>(cnt, 0, NN);
  edge_count<<<gEdge, 256, 0, stream>>>(dstI, cnt, (int)NE);
  dinv_from_cnt<<<gNode, 256, 0, stream>>>(cnt, dinv, NN);
  scan_tile_k<<<nTiles, 256, 0, stream>>>(cnt, rowptr, tileSum, (int)NN);
  scan_aux_k<<<1, 64, 0, stream>>>(tileSum, nTiles);
  scan_add_k<<<gNode, 256, 0, stream>>>(rowptr, tileSum, cursor, (int)NN, (int)NE);
  edge_place<<<gEdge, 256, 0, stream>>>(srcI, dstI, dinv, cursor, csr_src, csr_w, (int)NE);

  // ---- pack weights to bf16 WMMA-B fragment order
  pack_w<<<(DH * DH + 255) / 256, 256, 0, stream>>>(W1, Wp1, DH);
  pack_w<<<(DH * DH + 255) / 256, 256, 0, stream>>>(W2, Wp2, DH);
  pack_w<<<(DH * DH + 255) / 256, 256, 0, stream>>>(W3, Wp3, DH);
  pack_w<<<(DC * DH + 255) / 256, 256, 0, stream>>>(Wf1, Wpf1, DC);

  // ---- layer 1: x -> bufC (h1)
  gemm_wmma<DH><<<1024, 256, 0, stream>>>(x, Wp1, bufB, NN);
  aggregate_fused<<<gAgg, 256, 0, stream>>>(bufB, rowptr, csr_src, csr_w, dinv, b1, bufC, NN);
  bn_stats<<<DH, 256, 0, stream>>>(bufC, stats, NN);
  bn_apply<<<gN, 256, 0, stream>>>(bufC, stats, g1, be1, totN);

  // ---- layer 2: bufC -> bufA (h2)
  gemm_wmma<DH><<<1024, 256, 0, stream>>>(bufC, Wp2, bufB, NN);
  aggregate_fused<<<gAgg, 256, 0, stream>>>(bufB, rowptr, csr_src, csr_w, dinv, b2, bufA, NN);
  bn_stats<<<DH, 256, 0, stream>>>(bufA, stats, NN);
  bn_apply<<<gN, 256, 0, stream>>>(bufA, stats, g2, be2, totN);

  // ---- layer 3: bufA -> bufC (h3), no BN
  gemm_wmma<DH><<<1024, 256, 0, stream>>>(bufA, Wp3, bufB, NN);
  aggregate_fused<<<gAgg, 256, 0, stream>>>(bufB, rowptr, csr_src, csr_w, dinv, b3, bufC, NN);

  // ---- pooling + head
  fill_f<<<(int)((NG * DH + 255) / 256), 256, 0, stream>>>(hg, 0.0f, NG * DH);
  pool_add<<<gAgg, 256, 0, stream>>>(bufC, batch, hg, NN);
  concat_k<<<(int)((NG * DC + 255) / 256), 256, 0, stream>>>(hg, mol, hcat);
  gemm_wmma<DC><<<64, 256, 0, stream>>>(hcat, Wpf1, emb, NG);
  bias_selu<<<(int)((NG * DH + 255) / 256), 256, 0, stream>>>(emb, bf1, NG * DH);
  out_dot<<<(int)(NG / 2), 64, 0, stream>>>(emb, Wf2, bf2, out);
}
